// PointNet2FeatureExtractor_71605694759304
// MI455X (gfx1250) — matrix operand order repaired and checked by reading, not compiled, not run on hardware
//
#include <hip/hip_runtime.h>
#include <cstddef>

// ---------------------------------------------------------------------------
// PointNet++ (MSG) feature extractor for MI455X / gfx1250.
// All dense MLP/FC math goes through v_wmma_f32_16x16x32_bf16 (wave32 WMMA,
// f32 accumulate), with BN+bias+ReLU fused into the GEMM epilogue.
// ---------------------------------------------------------------------------

typedef __attribute__((ext_vector_type(16))) __bf16 v16bf;
typedef __attribute__((ext_vector_type(16))) short  short16;
typedef __attribute__((ext_vector_type(8)))  float  v8f;

#define NBATCH 8

__device__ __forceinline__ unsigned f2bf_lo(float x) {
  unsigned u = __builtin_bit_cast(unsigned, x);
  u = u + 0x7fffu + ((u >> 16) & 1u);          // round-to-nearest-even
  return u >> 16;
}
__device__ __forceinline__ unsigned pack2bf(float lo, float hi) {
  return f2bf_lo(lo) | (f2bf_lo(hi) << 16);
}

// ---------------------------------------------------------------------------
// Extract xyz (first 3 of 9 channels) from the raw pointcloud.
// ---------------------------------------------------------------------------
__global__ void k_extract_xyz(const float* __restrict__ pc, int n, float* __restrict__ xyz) {
  int i = blockIdx.x * blockDim.x + threadIdx.x;
  if (i >= n) return;
  xyz[i * 3 + 0] = pc[i * 9 + 0];
  xyz[i * 3 + 1] = pc[i * 9 + 1];
  xyz[i * 3 + 2] = pc[i * 9 + 2];
}

// ---------------------------------------------------------------------------
// Farthest point sampling. One 256-thread block per batch; running min-dist
// array lives entirely in LDS (N <= 4096 floats << 320KB/WGP).
// ---------------------------------------------------------------------------
__global__ __launch_bounds__(256) void k_fps(const float* __restrict__ xyz, int N,
                                             int npoint, int* __restrict__ fps_idx) {
  __shared__ float dist[4096];
  __shared__ float rv[256];
  __shared__ int   ri[256];
  __shared__ int   s_far;
  const int tid = threadIdx.x;
  const int b   = blockIdx.x;
  const float* xb = xyz + (size_t)b * N * 3;

  for (int j = tid; j < N; j += 256) dist[j] = 1e10f;
  __syncthreads();

  int far = 0;
  for (int i = 0; i < npoint; ++i) {
    if (tid == 0) fps_idx[b * npoint + i] = far;
    const float cx = xb[far * 3 + 0];
    const float cy = xb[far * 3 + 1];
    const float cz = xb[far * 3 + 2];
    float bd = -1.0f; int bj = 0;
    for (int j = tid; j < N; j += 256) {
      float dx = xb[j * 3 + 0] - cx;
      float dy = xb[j * 3 + 1] - cy;
      float dz = xb[j * 3 + 2] - cz;
      float d  = dx * dx + dy * dy + dz * dz;
      float nd = fminf(dist[j], d);
      dist[j]  = nd;
      if (nd > bd) { bd = nd; bj = j; }
    }
    rv[tid] = bd; ri[tid] = bj;
    __syncthreads();
    for (int s = 128; s > 0; s >>= 1) {
      if (tid < s && rv[tid + s] > rv[tid]) { rv[tid] = rv[tid + s]; ri[tid] = ri[tid + s]; }
      __syncthreads();
    }
    if (tid == 0) s_far = ri[0];
    __syncthreads();
    far = s_far;
  }
}

// ---------------------------------------------------------------------------
// Gather sampled centers: new_xyz[b,s] = xyz[b, fps_idx[b,s]].
// ---------------------------------------------------------------------------
__global__ void k_gather_xyz(const float* __restrict__ xyz, const int* __restrict__ idx,
                             int N, int S, int total, float* __restrict__ out) {
  int i = blockIdx.x * blockDim.x + threadIdx.x;
  if (i >= total) return;
  int b = i / S;
  int j = idx[i];
  out[i * 3 + 0] = xyz[((size_t)b * N + j) * 3 + 0];
  out[i * 3 + 1] = xyz[((size_t)b * N + j) * 3 + 1];
  out[i * 3 + 2] = xyz[((size_t)b * N + j) * 3 + 2];
}

// ---------------------------------------------------------------------------
// Ball query + gather + concat. One wave32 per query center. The reference's
// sort-based selection == "first K indices (ascending) within radius"; we
// reproduce that with a wave ballot + bit-scan (mask/cnt are wave-uniform).
// Writes g[center, k, 0:Cpts] = pts features, g[center, k, Cpts:Cpts+3] = dxyz.
// ---------------------------------------------------------------------------
__global__ __launch_bounds__(256) void k_query_group(
    const float* __restrict__ xyz, const float* __restrict__ nxyz,
    const float* __restrict__ pts, int N, int S, int K, int Cpts,
    float r2, float* __restrict__ g) {
  __shared__ int wls[8][32];
  const int tid  = threadIdx.x;
  const int lane = tid & 31;
  const int w    = tid >> 5;
  const int center = blockIdx.x * 8 + w;           // B*S is a multiple of 8
  const int b = center / S;

  const float qx = nxyz[center * 3 + 0];
  const float qy = nxyz[center * 3 + 1];
  const float qz = nxyz[center * 3 + 2];

  int cnt = 0, first = -1;
  for (int base = 0; base < N && cnt < K; base += 32) {   // N is a multiple of 32
    int j = base + lane;
    float dx = xyz[((size_t)b * N + j) * 3 + 0] - qx;
    float dy = xyz[((size_t)b * N + j) * 3 + 1] - qy;
    float dz = xyz[((size_t)b * N + j) * 3 + 2] - qz;
    bool pred = (dx * dx + dy * dy + dz * dz) <= r2;
    unsigned mask = (unsigned)__ballot(pred);             // wave32: low 32 bits
    while (mask && cnt < K) {
      int bit = __ffs(mask) - 1;
      if (lane == 0) wls[w][cnt] = base + bit;
      if (first < 0) first = base + bit;
      ++cnt;
      mask &= mask - 1;
    }
  }
  if (cnt == 0) first = N - 1;                            // reference min(n, n-1) fallback
  if (lane == 0) for (int kk = cnt; kk < K; ++kk) wls[w][kk] = first;
  __syncthreads();

  const int Cin = Cpts + 3;
  for (int kk = 0; kk < K; ++kk) {
    int j = wls[w][kk];
    const float* prow = pts + ((size_t)b * N + j) * Cpts;
    float gx = xyz[((size_t)b * N + j) * 3 + 0] - qx;
    float gy = xyz[((size_t)b * N + j) * 3 + 1] - qy;
    float gz = xyz[((size_t)b * N + j) * 3 + 2] - qz;
    size_t orow = ((size_t)center * K + kk) * Cin;
    for (int c = lane; c < Cin; c += 32) {
      float v;
      if (c < Cpts)           v = prow[c];
      else if (c == Cpts)     v = gx;
      else if (c == Cpts + 1) v = gy;
      else                    v = gz;
      g[orow + c] = v;
    }
  }
}

// ---------------------------------------------------------------------------
// Fold BN into per-channel scale/shift:  y = acc*s + t
//   s = gamma * rsqrt(var+eps), t = (bias-mean)*s + beta   (or s=1,t=bias)
// ---------------------------------------------------------------------------
__global__ void k_prep_st(const float* __restrict__ bias, const float* __restrict__ gamma,
                          const float* __restrict__ beta, const float* __restrict__ mean,
                          const float* __restrict__ var, int C,
                          float* __restrict__ s, float* __restrict__ t) {
  int c = blockIdx.x * blockDim.x + threadIdx.x;
  if (c >= C) return;
  if (gamma) {
    float sc = gamma[c] * rsqrtf(var[c] + 1e-5f);
    s[c] = sc;
    t[c] = (bias[c] - mean[c]) * sc + beta[c];
  } else {
    s[c] = 1.0f;
    t[c] = bias[c];
  }
}

// ---------------------------------------------------------------------------
// WMMA GEMM:  Out[M,Cout] = act( (A[M,Cin] @ W[Cout,Cin]^T) * s + t )
// Block = 128 threads = 4 waves; block tile = 128 rows x 16 cols. Each wave
// owns TWO 16x16 M-tiles that share one B fragment -> 2x
// v_wmma_f32_16x16x32_bf16 per K-step.
//
// Staging: every thread converts two adjacent K elements and stores ONE
// packed uint32 to LDS (ds_store_b32, 4B-aligned because columns are even and
// the row stride is 40 halfwords) — no sub-word LDS stores, no exec-mask
// branches. A wave-uniform "interior" test skips all clamps/selects for full
// tiles (M is a multiple of 128 for every MLP GEMM), so only ragged Cin/Cout
// edge tiles pay the clamped path.
// Fragment layouts per CDNA5 ISA §7.12.2.
// ---------------------------------------------------------------------------
__global__ __launch_bounds__(128) void k_gemm_bn_act(
    const float* __restrict__ A, int M, int Cin,
    const float* __restrict__ W, int Cout,
    const float* __restrict__ sc, const float* __restrict__ sh,
    int relu, float* __restrict__ Out) {
  __shared__ unsigned short lA[128][40];  // 128 rows x 32 bf16 (+pad)
  __shared__ unsigned short lW[16][40];   // 16 out-chans x 32 bf16 (+pad)
  const int tid  = threadIdx.x;
  const int lane = tid & 31;
  const int wave = tid >> 5;
  const int m0 = blockIdx.x * 128;
  const int n0 = blockIdx.y * 16;

  v8f acc0 = {};
  v8f acc1 = {};

  for (int k0 = 0; k0 < Cin; k0 += 32) {
    const bool kfull = (k0 + 32 <= Cin);          // uniform
    const bool afull = kfull && (m0 + 128 <= M);  // uniform
    const bool wfull = kfull && (n0 + 16 <= Cout);// uniform

    // ---- Stage weight tile (16x32 bf16 = 256 pairs): 128 threads x 2 pairs.
    for (int e = tid; e < 256; e += 128) {
      int r  = e >> 4;                 // out-channel row 0..15
      int pc = (e & 15) * 2;           // even column
      int o  = n0 + r;
      unsigned pk;
      if (wfull) {
        const float* p = &W[(size_t)o * Cin + k0 + pc];
        pk = pack2bf(p[0], p[1]);
      } else {
        int orow = (o < Cout) ? o : (Cout - 1);
        int c0 = k0 + pc, c1 = c0 + 1;
        int c0c = (c0 < Cin) ? c0 : (Cin - 1);
        int c1c = (c1 < Cin) ? c1 : (Cin - 1);
        float v0 = W[(size_t)orow * Cin + c0c];
        float v1 = W[(size_t)orow * Cin + c1c];
        v0 = (o < Cout && c0 < Cin) ? v0 : 0.0f;
        v1 = (o < Cout && c1 < Cin) ? v1 : 0.0f;
        pk = pack2bf(v0, v1);
      }
      *(unsigned*)&lW[r][pc] = pk;
    }

    // ---- Stage activation tile (128x32 bf16 = 2048 pairs): 128 x 16 pairs.
    for (int e = tid; e < 2048; e += 128) {
      int r  = e >> 4;                 // row 0..127
      int pc = (e & 15) * 2;           // even column
      int m  = m0 + r;
      unsigned pk;
      if (afull) {
        const float* p = &A[(size_t)m * Cin + k0 + pc];
        pk = pack2bf(p[0], p[1]);
      } else {
        int mm = (m < M) ? m : (M - 1);
        int c0 = k0 + pc, c1 = c0 + 1;
        int c0c = (c0 < Cin) ? c0 : (Cin - 1);
        int c1c = (c1 < Cin) ? c1 : (Cin - 1);
        float v0 = A[(size_t)mm * Cin + c0c];
        float v1 = A[(size_t)mm * Cin + c1c];
        v0 = (m < M && c0 < Cin) ? v0 : 0.0f;
        v1 = (m < M && c1 < Cin) ? v1 : 0.0f;
        pk = pack2bf(v0, v1);
      }
      *(unsigned*)&lA[r][pc] = pk;
    }

    // ---- One uniform prefetch of the next K tile per thread.
    if (k0 + 32 < Cin) {
      int m = m0 + tid;
      int mm = (m < M) ? m : (M - 1);
      __builtin_prefetch(&A[(size_t)mm * Cin + k0 + 32], 0, 1);  // global_prefetch_b8
    }
    __syncthreads();

    // A fragments (16x32 bf16): lanes 0-15 -> K{0..7,16..23}; 16-31 -> K{8..15,24..31}
    const int mrow0 = wave * 32 + (lane & 15);
    const int mrow1 = mrow0 + 16;
    const int khi   = (lane >> 4) * 8;
    short16 av0, av1, bv;
    for (int i = 0; i < 8; ++i) {
      av0[i]     = (short)lA[mrow0][khi + i];
      av0[i + 8] = (short)lA[mrow0][16 + khi + i];
      av1[i]     = (short)lA[mrow1][khi + i];
      av1[i + 8] = (short)lA[mrow1][16 + khi + i];
    }
    // B fragment (32x16 bf16): B[k][n] = W[n][k]; lanes 0-15 -> K{0..7,16..23}; 16-31 -> K{8..15,24..31}
    const int ncol = lane & 15;
    const int kb   = (lane >> 4) * 8;
    for (int i = 0; i < 8; ++i) {
      bv[i]     = (short)lW[ncol][kb + i];
      bv[i + 8] = (short)lW[ncol][16 + kb + i];
    }

    v16bf af0 = __builtin_bit_cast(v16bf, av0);
    v16bf af1 = __builtin_bit_cast(v16bf, av1);
    v16bf bf  = __builtin_bit_cast(v16bf, bv);
    acc0 = __builtin_amdgcn_wmma_f32_16x16x32_bf16(false, af0, false, bf,
                                                   (short)0, acc0, false, false);
    acc1 = __builtin_amdgcn_wmma_f32_16x16x32_bf16(false, af1, false, bf,
                                                   (short)0, acc1, false, false);
    __syncthreads();
  }

  // Epilogue: C/D layout — VGPR r: lanes 0-15 => M=r, lanes 16-31 => M=8+r; N=lane&15.
  const int nn = n0 + (lane & 15);
  const float s = (nn < Cout) ? sc[nn] : 0.0f;
  const float t = (nn < Cout) ? sh[nn] : 0.0f;
  for (int r = 0; r < 8; ++r) {
    int mmb = m0 + wave * 32 + (lane >> 4) * 8 + r;
    if (mmb < M && nn < Cout) {
      float y = acc0[r] * s + t;
      if (relu) y = fmaxf(y, 0.0f);
      Out[(size_t)mmb * Cout + nn] = y;
    }
    int mm1 = mmb + 16;
    if (mm1 < M && nn < Cout) {
      float y = acc1[r] * s + t;
      if (relu) y = fmaxf(y, 0.0f);
      Out[(size_t)mm1 * Cout + nn] = y;
    }
  }
}

// ---------------------------------------------------------------------------
// Max over K neighbors: out[b,s, coff+c] = max_k in[(b*S+s)*K + k, c]
// ---------------------------------------------------------------------------
__global__ void k_maxpool_k(const float* __restrict__ in, int BS, int K, int C,
                            float* __restrict__ out, int Ctot, int coff) {
  int i = blockIdx.x * blockDim.x + threadIdx.x;
  if (i >= BS * C) return;
  int p = i / C, c = i % C;
  const float* row = in + (size_t)p * K * C + c;
  float m = row[0];
  for (int kk = 1; kk < K; ++kk) m = fmaxf(m, row[(size_t)kk * C]);
  out[(size_t)p * Ctot + coff + c] = m;
}

// ---------------------------------------------------------------------------
// Global max over points: out[b,c] = max_s in[b,s,c]
// ---------------------------------------------------------------------------
__global__ void k_maxpool_s(const float* __restrict__ in, int S, int C, int total,
                            float* __restrict__ out) {
  int i = blockIdx.x * blockDim.x + threadIdx.x;
  if (i >= total) return;
  int b = i / C, c = i % C;
  float m = in[(size_t)b * S * C + c];
  for (int s = 1; s < S; ++s) m = fmaxf(m, in[((size_t)b * S + s) * C + c]);
  out[i] = m;
}

// ---------------------------------------------------------------------------
// Host orchestration.
// ---------------------------------------------------------------------------
struct BrCfg { float r; int K; int ch[3]; };
struct StCfg { int N, S, Cpts, Ctot; BrCfg br[2]; };

static const StCfg ST[3] = {
  {4096, 1024,   9,  96, {{0.05f, 16, {16, 16, 32}},   {0.10f, 32, {32, 32, 64}}}},
  {1024,  256,  96, 256, {{0.10f, 16, {64, 64, 128}},  {0.20f, 32, {64, 96, 128}}}},
  { 256,   64, 256, 512, {{0.20f, 16, {128, 196, 256}},{0.40f, 32, {128, 196, 256}}}},
};

static inline int cdiv(int a, int b) { return (a + b - 1) / b; }

extern "C" void kernel_launch(void* const* d_in, const int* in_sizes, int n_in,
                              void* d_out, int out_size, void* d_ws, size_t ws_size,
                              hipStream_t stream) {
  (void)in_sizes; (void)n_in; (void)out_size;

  // d_in layout (setup_inputs insertion order):
  //  [0] pointcloud; [1..108] 3 stages x 2 branches x 3 layers x {W,b,gamma,beta,mean,var};
  //  [109] fc1_W [110] fc1_b [111] bn1_gamma [112] bn1_beta [113] bn1_mean [114] bn1_var
  //  [115] fc2_W [116] fc2_b
  auto P  = [&](int i) { return (const float*)d_in[i]; };
  auto li = [&](int st, int br, int ly) { return 1 + st * 36 + br * 18 + ly * 6; };

  // Bump allocator over d_ws (~133 MB total).
  size_t off = 0;
  auto alloc = [&](size_t bytes) -> void* {
    void* p = (char*)d_ws + off;
    off = (off + bytes + 255) & ~(size_t)255;
    return p;
  };
  float* xyz0  = (float*)alloc((size_t)NBATCH * 4096 * 3 * 4);
  int*   fpsI  = (int*)  alloc((size_t)NBATCH * 1024 * 4);
  float* nx[3];
  nx[0] = (float*)alloc((size_t)NBATCH * 1024 * 3 * 4);
  nx[1] = (float*)alloc((size_t)NBATCH *  256 * 3 * 4);
  nx[2] = (float*)alloc((size_t)NBATCH *   64 * 3 * 4);
  float* ptsb[3];
  ptsb[0] = (float*)alloc((size_t)NBATCH * 1024 *  96 * 4);
  ptsb[1] = (float*)alloc((size_t)NBATCH *  256 * 256 * 4);
  ptsb[2] = (float*)alloc((size_t)NBATCH *   64 * 512 * 4);
  float* g    = (float*)alloc((size_t)6488064 * 4);   // max B*S*K*(Cpts+3)
  float* ping = (float*)alloc((size_t)16777216 * 4);  // max M*Cout
  float* pong = (float*)alloc((size_t)8388608 * 4);
  float* sbuf = (float*)alloc(512 * 4);
  float* tbuf = (float*)alloc(512 * 4);
  float* xmax = (float*)alloc((size_t)NBATCH * 512 * 4);
  float* fc1o = (float*)alloc((size_t)NBATCH * 512 * 4);
  (void)ws_size;

  const float* pc = P(0);
  k_extract_xyz<<<cdiv(NBATCH * 4096, 256), 256, 0, stream>>>(pc, NBATCH * 4096, xyz0);

  const float* xyz  = xyz0;
  const float* pts  = pc;
  for (int st = 0; st < 3; ++st) {
    const StCfg& c = ST[st];
    // FPS + center gather
    k_fps<<<NBATCH, 256, 0, stream>>>(xyz, c.N, c.S, fpsI);
    k_gather_xyz<<<cdiv(NBATCH * c.S, 256), 256, 0, stream>>>(xyz, fpsI, c.N, c.S,
                                                              NBATCH * c.S, nx[st]);
    int coff = 0;
    for (int br = 0; br < 2; ++br) {
      const BrCfg& bc = c.br[br];
      const int Cin0 = c.Cpts + 3;
      const int M    = NBATCH * c.S * bc.K;
      k_query_group<<<NBATCH * c.S / 8, 256, 0, stream>>>(
          xyz, nx[st], pts, c.N, c.S, bc.K, c.Cpts, bc.r * bc.r, g);

      const float* Ain = g;
      int cin = Cin0;
      float* Aout = ping;
      for (int ly = 0; ly < 3; ++ly) {
        const int base = li(st, br, ly);
        const int cout = bc.ch[ly];
        k_prep_st<<<cdiv(cout, 128), 128, 0, stream>>>(
            P(base + 1), P(base + 2), P(base + 3), P(base + 4), P(base + 5),
            cout, sbuf, tbuf);
        Aout = (ly == 1) ? pong : ping;   // g->ping->pong->ping
        dim3 grid(cdiv(M, 128), cdiv(cout, 16));
        k_gemm_bn_act<<<grid, 128, 0, stream>>>(Ain, M, cin, P(base), cout,
                                                sbuf, tbuf, 1, Aout);
        Ain = Aout;
        cin = cout;
      }
      k_maxpool_k<<<cdiv(NBATCH * c.S * cin, 256), 256, 0, stream>>>(
          Ain, NBATCH * c.S, bc.K, cin, ptsb[st], c.Ctot, coff);
      coff += cin;
    }
    xyz = nx[st];
    pts = ptsb[st];
  }

  // Head: global max -> fc1(+bn+relu) -> fc2
  k_maxpool_s<<<cdiv(NBATCH * 512, 256), 256, 0, stream>>>(ptsb[2], 64, 512,
                                                           NBATCH * 512, xmax);
  const int FC = 1 + 3 * 36;  // 109
  k_prep_st<<<cdiv(512, 128), 128, 0, stream>>>(P(FC + 1), P(FC + 2), P(FC + 3),
                                                P(FC + 4), P(FC + 5), 512, sbuf, tbuf);
  {
    dim3 grid(cdiv(NBATCH, 128), cdiv(512, 16));
    k_gemm_bn_act<<<grid, 128, 0, stream>>>(xmax, NBATCH, 512, P(FC), 512,
                                            sbuf, tbuf, 1, fc1o);
  }
  k_prep_st<<<cdiv(248, 128), 128, 0, stream>>>(P(FC + 7), nullptr, nullptr,
                                                nullptr, nullptr, 248, sbuf, tbuf);
  {
    dim3 grid(cdiv(NBATCH, 128), cdiv(248, 16));
    k_gemm_bn_act<<<grid, 128, 0, stream>>>(fc1o, NBATCH, 512, P(FC + 6), 248,
                                            sbuf, tbuf, 0, (float*)d_out);
  }
}
